// KANSpikingNeuron_54778012893659
// MI455X (gfx1250) — matrix operand order repaired and checked by reading, not compiled, not run on hardware
//
#include <hip/hip_runtime.h>
#include <stdint.h>

typedef __bf16 v16bf __attribute__((ext_vector_type(16)));
typedef __bf16 v8bf  __attribute__((ext_vector_type(8)));
typedef __bf16 v4bf  __attribute__((ext_vector_type(4)));
typedef float  v8f   __attribute__((ext_vector_type(8)));

#define IN_DIM  4096
#define HID_DIM 4096
#define OUT_DIM 4096
#define BATCH   8192

#define BM 128
#define BN 128
#define BK 32
#define LDS_PITCH 40   // BK + 8 bf16 pad: row pitch 80B stays 16B-aligned

// ---------- fp32 -> bf16 conversion, 4 elems/thread ----------
__global__ __launch_bounds__(256) void cvt_f32_to_bf16(const float* __restrict__ in,
                                                       __bf16* __restrict__ out, int n4) {
  int i = blockIdx.x * 256 + threadIdx.x;
  if (i < n4) {
    float4 f = ((const float4*)in)[i];
    v4bf o = { (__bf16)f.x, (__bf16)f.y, (__bf16)f.z, (__bf16)f.w };
    ((v4bf*)out)[i] = o;
  }
}

// ---------- bf16 GEMM: C = A[M,K] * W[N,K]^T, fused bias + activation ----------
// EPI == 0: out = bf16( relu(c + bias) )   (layer 1 -> h)
// EPI == 1: out = float( (c + bias) > 0 )  (layer 2 -> spikes)
template <int EPI>
__global__ __launch_bounds__(256) void gemm_bias_act(
    const __bf16* __restrict__ A,    // [M][K] bf16
    const __bf16* __restrict__ W,    // [N][K] bf16 (row-major == B^T)
    const float*  __restrict__ bias, // [N]
    void* __restrict__ outp,
    int M, int N, int K)
{
  __shared__ __bf16 As[2][BM][LDS_PITCH];
  __shared__ __bf16 Bs[2][BM][LDS_PITCH];

  const int tid   = threadIdx.x;
  const int wave  = tid >> 5;
  const int lane  = tid & 31;
  const int half  = lane >> 4;   // wave32: two 16-lane halves
  const int l15   = lane & 15;
  const int waveM = wave >> 2;   // 0..1  (64-row slab)
  const int waveN = wave & 3;    // 0..3  (32-col slab)
  const int bm = blockIdx.y * BM;
  const int bn = blockIdx.x * BN;

  const unsigned asBase   = (unsigned)(uintptr_t)&As[0][0][0];
  const unsigned bsBase   = (unsigned)(uintptr_t)&Bs[0][0][0];
  const unsigned bufBytes = BM * LDS_PITCH * 2;

  // Stage one 128x32 bf16 tile of A and of W into LDS buffer `buf`
  // via CDNA5 async global->LDS copies (ASYNCcnt-tracked).
  auto stage = [&](int buf, int k0) {
    const __bf16* ga0 = A + (size_t)bm * K + k0;
    const __bf16* gb0 = W + (size_t)bn * K + k0;
    unsigned aL = asBase + (unsigned)buf * bufBytes;
    unsigned bL = bsBase + (unsigned)buf * bufBytes;
#pragma unroll
    for (int p = 0; p < 2; ++p) {
      int chunk = tid + p * 256;          // 512 x 16B chunks per tile
      int row   = chunk >> 2;             // 0..127
      int col   = (chunk & 3) * 8;        // 0,8,16,24 (bf16 elems)
      unsigned lofs = (unsigned)(row * LDS_PITCH + col) * 2u;
      unsigned long long gA = (unsigned long long)(ga0 + (size_t)row * K + col);
      unsigned long long gB = (unsigned long long)(gb0 + (size_t)row * K + col);
      unsigned la = aL + lofs, lb = bL + lofs;
      asm volatile("global_load_async_to_lds_b128 %0, %1, off"
                   :: "v"(la), "v"(gA) : "memory");
      asm volatile("global_load_async_to_lds_b128 %0, %1, off"
                   :: "v"(lb), "v"(gB) : "memory");
    }
  };

  v8f acc[4][2] = {};

  stage(0, 0);
  asm volatile("s_wait_asynccnt 0x0" ::: "memory");
  __syncthreads();

  const int nk = K / BK;
  int buf = 0;
  for (int kt = 0; kt < nk; ++kt) {
    if (kt + 1 < nk) stage(buf ^ 1, (kt + 1) * BK);

    union Frag { v16bf v; v8bf h[2]; };
    Frag a[4], b[2];
    // A fragment (16x32 bf16): lane half selects K-groups {0..7,16..23} / {8..15,24..31}
#pragma unroll
    for (int mt = 0; mt < 4; ++mt) {
      const __bf16* r = &As[buf][waveM * 64 + mt * 16 + l15][0];
      a[mt].h[0] = *(const v8bf*)(r + 8 * half);
      a[mt].h[1] = *(const v8bf*)(r + 16 + 8 * half);
    }
    // B fragment (32x16 bf16): lane half selects K-groups {0..7,16..23} / {8..15,24..31}
#pragma unroll
    for (int nt = 0; nt < 2; ++nt) {
      const __bf16* r = &Bs[buf][waveN * 32 + nt * 16 + l15][0];
      b[nt].h[0] = *(const v8bf*)(r + 8 * half);
      b[nt].h[1] = *(const v8bf*)(r + 16 + 8 * half);
    }
#pragma unroll
    for (int mt = 0; mt < 4; ++mt)
#pragma unroll
      for (int nt = 0; nt < 2; ++nt)
        acc[mt][nt] = __builtin_amdgcn_wmma_f32_16x16x32_bf16(
            false, a[mt].v, false, b[nt].v, (short)0, acc[mt][nt], false, false);

    asm volatile("s_wait_asynccnt 0x0" ::: "memory");
    __syncthreads();
    buf ^= 1;
  }

  // Epilogue: C/D f32 layout — lane N=l15, VGPR r -> row r + 8*half within 16x16 tile
#pragma unroll
  for (int nt = 0; nt < 2; ++nt) {
    int n = bn + waveN * 32 + nt * 16 + l15;
    float bv = bias[n];
#pragma unroll
    for (int mt = 0; mt < 4; ++mt) {
#pragma unroll
      for (int r = 0; r < 8; ++r) {
        int m = bm + waveM * 64 + mt * 16 + 8 * half + r;
        float y = acc[mt][nt][r] + bv;
        if (EPI == 0) {
          ((__bf16*)outp)[(size_t)m * N + n] = (__bf16)fmaxf(y, 0.0f);
        } else {
          ((float*)outp)[(size_t)m * N + n] = (y > 0.0f) ? 1.0f : 0.0f;
        }
      }
    }
  }
}

extern "C" void kernel_launch(void* const* d_in, const int* in_sizes, int n_in,
                              void* d_out, int out_size, void* d_ws, size_t ws_size,
                              hipStream_t stream) {
  const float* x  = (const float*)d_in[0];
  const float* W1 = (const float*)d_in[1];
  const float* b1 = (const float*)d_in[2];
  const float* W2 = (const float*)d_in[3];
  const float* b2 = (const float*)d_in[4];

  char* ws = (char*)d_ws;
  const size_t xbBytes = (size_t)BATCH * IN_DIM * 2;
  const size_t wBytes  = (size_t)HID_DIM * IN_DIM * 2;
  __bf16* xb  = (__bf16*)(ws);
  __bf16* w1b = (__bf16*)(ws + xbBytes);
  __bf16* w2b = (__bf16*)(ws + xbBytes + wBytes);
  __bf16* h   = (__bf16*)(ws + xbBytes + 2 * wBytes);

  const int nx = BATCH * IN_DIM / 4;     // 4 elems per thread
  const int nw = HID_DIM * IN_DIM / 4;
  cvt_f32_to_bf16<<<nx / 256, 256, 0, stream>>>(x,  xb,  nx);
  cvt_f32_to_bf16<<<nw / 256, 256, 0, stream>>>(W1, w1b, nw);
  cvt_f32_to_bf16<<<nw / 256, 256, 0, stream>>>(W2, w2b, nw);

  dim3 block(256);
  dim3 grid1(HID_DIM / BN, BATCH / BM);
  gemm_bias_act<0><<<grid1, block, 0, stream>>>(xb, w1b, b1, (void*)h,
                                                BATCH, HID_DIM, IN_DIM);
  dim3 grid2(OUT_DIM / BN, BATCH / BM);
  gemm_bias_act<1><<<grid2, block, 0, stream>>>(h, w2b, b2, d_out,
                                                BATCH, OUT_DIM, HID_DIM);
}
